// TransformerModel_61194694033510
// MI455X (gfx1250) — compile-verified, matrix operand order checked by reading, not run on hardware
//
#include <hip/hip_runtime.h>
#include <hip/hip_bf16.h>
#include <math.h>

// ---------------------------------------------------------------------------
// GPT2-ish forward (B=8,N=512,NDIMS=64,D=1024,L=8,H=16,HD=64,DFF=4096), fp32
// I/O, f16 WMMA compute (v_wmma_f32_16x16x32_f16) with fp32 accumulation.
// Weights stay fp32 in HBM (each layer's ~50MB fits the 192MB L2) and are
// converted to f16 during the global->LDS stage of each GEMM.
//
// GEMM: block tile 128x128, BK=32, 8 waves in a 2(M)x4(N) grid; each wave
// owns a 64x32 region (4x2 WMMA tiles -> 8 v_wmma per K-step, 64 VGPR accum).
// Double-buffered LDS + register prefetch: one barrier per K-step, global
// loads of tile k+1 overlap WMMA of tile k.
// ---------------------------------------------------------------------------

typedef _Float16 half_t;
typedef _Float16 v8h  __attribute__((ext_vector_type(8)));
typedef _Float16 v16h __attribute__((ext_vector_type(16)));
typedef float    v8f  __attribute__((ext_vector_type(8)));

#define NTOK   4096          // B*N
#define DMODEL 1024
#define SEQ    512
#define NHEAD  16
#define HD     64
#define DFF    4096

// Build a 16x(K=32) WMMA operand fragment for this lane from an LDS row.
// Layout per CDNA5 ISA 7.12.2 (16-bit A 16x32): lanes 0-15 take k-halves
// {0..7, 16..23}, lanes 16-31 take {8..15, 24..31}.  rowp points at the
// lane's row (A: row m, B: column n of the transposed-stored tile).
__device__ inline v16h frag_ld(const half_t* rowp, int hi8) {
  v8h lo = *(const v8h*)(rowp + hi8);
  v8h hh = *(const v8h*)(rowp + 16 + hi8);
  return __builtin_shufflevector(lo, hh, 0,1,2,3,4,5,6,7,8,9,10,11,12,13,14,15);
}

__device__ inline v8f wmma_f16(v16h a, v16h b, v8f c) {
  return __builtin_amdgcn_wmma_f32_16x16x32_f16(
      /*neg_a=*/false, a, /*neg_b=*/false, b,
      /*c_mod=*/(short)0, c, /*reuse_a=*/false, /*reuse_b=*/false);
}

__device__ inline float gelu_new(float x) {
  const float c0 = 0.7978845608028654f;   // sqrt(2/pi)
  return 0.5f * x * (1.0f + tanhf(c0 * (x + 0.044715f * x * x * x)));
}

// ---------------------------------------------------------------------------
// Embedding: h[t][c] = sum_k xs[t][k]*Wri[k][c] + bri[c] + wpe[t%512][c]
// K=64 only -> plain VALU kernel; negligible fraction of total FLOPs.
// ---------------------------------------------------------------------------
__global__ __launch_bounds__(256) void embed_k(
    const float* __restrict__ xs, const float* __restrict__ Wri,
    const float* __restrict__ bri, const float* __restrict__ wpe,
    float* __restrict__ h) {
  __shared__ float sx[64];
  int t = blockIdx.x, n = t & (SEQ - 1), tid = threadIdx.x;
  if (tid < 64) sx[tid] = xs[(size_t)t * 64 + tid];
  __syncthreads();
#pragma unroll
  for (int c = 0; c < 4; ++c) {
    int col = tid + c * 256;
    float acc = bri[col] + wpe[(size_t)n * DMODEL + col];
#pragma unroll 8
    for (int k = 0; k < 64; ++k) acc += sx[k] * Wri[(size_t)k * DMODEL + col];
    h[(size_t)t * DMODEL + col] = acc;
  }
}

// ---------------------------------------------------------------------------
// LayerNorm over D=1024, one block per token.  OUT = half_t (feed GEMMs) or
// float (final LN before the head).
// ---------------------------------------------------------------------------
template <typename OUT>
__global__ __launch_bounds__(256) void layernorm_k(
    const float* __restrict__ x, const float* __restrict__ g,
    const float* __restrict__ b, OUT* __restrict__ y) {
  __shared__ float red[256];
  int row = blockIdx.x, tid = threadIdx.x;
  const float* xr = x + (size_t)row * DMODEL;
  float v[4], s = 0.f;
#pragma unroll
  for (int i = 0; i < 4; ++i) { v[i] = xr[tid + i * 256]; s += v[i]; }
  red[tid] = s; __syncthreads();
  for (int off = 128; off; off >>= 1) {
    if (tid < off) red[tid] += red[tid + off];
    __syncthreads();
  }
  float mu = red[0] * (1.0f / DMODEL);
  __syncthreads();
  float s2 = 0.f;
#pragma unroll
  for (int i = 0; i < 4; ++i) { float d = v[i] - mu; s2 += d * d; }
  red[tid] = s2; __syncthreads();
  for (int off = 128; off; off >>= 1) {
    if (tid < off) red[tid] += red[tid + off];
    __syncthreads();
  }
  float rstd = rsqrtf(red[0] * (1.0f / DMODEL) + 1e-5f);
#pragma unroll
  for (int i = 0; i < 4; ++i) {
    int col = tid + i * 256;
    y[(size_t)row * DMODEL + col] = (OUT)((v[i] - mu) * rstd * g[col] + b[col]);
  }
}

// ---------------------------------------------------------------------------
// WMMA GEMM:  C[M,N] = A16[M,K] @ W32[K,N] + bias, with epilogue:
//   EPI==0: store f16          EPI==1: gelu_new then store f16
//   EPI==2: resid32 += C       (in-place fp32 residual add)
// ---------------------------------------------------------------------------
#define LPITCH 40                      // 32 + 8 pad halves (80B rows, 16B ok)
#define LBUF   (128 * LPITCH)          // one 128x32 tile

template <int EPI>
__global__ __launch_bounds__(256) void gemm_ws(
    const half_t* __restrict__ A, const float* __restrict__ Bw,
    const float* __restrict__ bias, float* __restrict__ resid32,
    half_t* __restrict__ out16, int M, int N, int K) {
  __shared__ half_t As[2 * LBUF];      // [buf][m][k]
  __shared__ half_t Bs[2 * LBUF];      // [buf][n][k] (transposed W tile)

  int tid = threadIdx.x, lane = tid & 31, w = tid >> 5;
  int wm = w & 1, wn = w >> 1;               // wave grid 2(M) x 4(N)
  int r = lane & 15, hi8 = (lane >> 4) << 3; // hi8 in {0,8}

  int m0 = blockIdx.y * 128, n0 = blockIdx.x * 128;

  // Per-thread global-load coordinates (fixed across K-steps).
  int arow = tid >> 2, acol = (tid & 3) << 3;   // A: 2 chunks, rows arow,arow+64
  int bn = (tid & 31) << 2, kb = tid >> 5;      // B: 4x float4, k = kb + 8i

  const v8f vzero = {0.f, 0.f, 0.f, 0.f, 0.f, 0.f, 0.f, 0.f};
  v8f acc[4][2];
#pragma unroll
  for (int mi = 0; mi < 4; ++mi) { acc[mi][0] = vzero; acc[mi][1] = vzero; }

  v8h areg[2];
  float4 breg[4];

  // ---- prologue: fetch tile 0 ----
#pragma unroll
  for (int c = 0; c < 2; ++c)
    areg[c] = *(const v8h*)(A + (size_t)(m0 + arow + 64 * c) * K + acol);
#pragma unroll
  for (int i = 0; i < 4; ++i)
    breg[i] = *(const float4*)(Bw + (size_t)(kb + (i << 3)) * N + n0 + bn);
#pragma unroll
  for (int c = 0; c < 2; ++c)
    *(v8h*)&As[(arow + 64 * c) * LPITCH + acol] = areg[c];
#pragma unroll
  for (int i = 0; i < 4; ++i) {
    int bk = kb + (i << 3);
    Bs[(bn + 0) * LPITCH + bk] = (half_t)breg[i].x;
    Bs[(bn + 1) * LPITCH + bk] = (half_t)breg[i].y;
    Bs[(bn + 2) * LPITCH + bk] = (half_t)breg[i].z;
    Bs[(bn + 3) * LPITCH + bk] = (half_t)breg[i].w;
  }
  __syncthreads();

  int kIter = K >> 5, buf = 0;
  for (int it = 0; it < kIter; ++it) {
    // ---- prefetch tile it+1 into registers (overlaps WMMA below) ----
    if (it + 1 < kIter) {
      int kk = (it + 1) << 5;
#pragma unroll
      for (int c = 0; c < 2; ++c)
        areg[c] = *(const v8h*)(A + (size_t)(m0 + arow + 64 * c) * K + kk + acol);
#pragma unroll
      for (int i = 0; i < 4; ++i)
        breg[i] = *(const float4*)(Bw + (size_t)(kk + kb + (i << 3)) * N + n0 + bn);
    }

    // ---- compute from LDS[buf]: 6 fragment loads -> 8 WMMAs ----
    {
      const half_t* Ab = &As[buf * LBUF];
      const half_t* Bb = &Bs[buf * LBUF];
      v16h bf0 = frag_ld(&Bb[(wn * 32 + r) * LPITCH], hi8);
      v16h bf1 = frag_ld(&Bb[(wn * 32 + 16 + r) * LPITCH], hi8);
#pragma unroll
      for (int mi = 0; mi < 4; ++mi) {
        v16h af = frag_ld(&Ab[(wm * 64 + mi * 16 + r) * LPITCH], hi8);
        acc[mi][0] = wmma_f16(af, bf0, acc[mi][0]);
        acc[mi][1] = wmma_f16(af, bf1, acc[mi][1]);
      }
    }

    // ---- drain registers into the other LDS buffer, single barrier ----
    if (it + 1 < kIter) {
      int nb = buf ^ 1;
      half_t* Ad = &As[nb * LBUF];
      half_t* Bd = &Bs[nb * LBUF];
#pragma unroll
      for (int c = 0; c < 2; ++c)
        *(v8h*)&Ad[(arow + 64 * c) * LPITCH + acol] = areg[c];
#pragma unroll
      for (int i = 0; i < 4; ++i) {
        int bk = kb + (i << 3);
        Bd[(bn + 0) * LPITCH + bk] = (half_t)breg[i].x;
        Bd[(bn + 1) * LPITCH + bk] = (half_t)breg[i].y;
        Bd[(bn + 2) * LPITCH + bk] = (half_t)breg[i].z;
        Bd[(bn + 3) * LPITCH + bk] = (half_t)breg[i].w;
      }
      __syncthreads();
      buf = nb;
    }
  }

  // Epilogue.  C layout: lane l, vgpr v -> (m = v + 8*(l>=16), n = l%16).
#pragma unroll
  for (int mi = 0; mi < 4; ++mi) {
#pragma unroll
    for (int ni = 0; ni < 2; ++ni) {
      int col = n0 + wn * 32 + ni * 16 + r;
      float bv = bias[col];
#pragma unroll
      for (int v = 0; v < 8; ++v) {
        int row = m0 + wm * 64 + mi * 16 + v + hi8;
        float val = acc[mi][ni][v] + bv;
        if constexpr (EPI == 0) {
          out16[(size_t)row * N + col] = (half_t)val;
        } else if constexpr (EPI == 1) {
          out16[(size_t)row * N + col] = (half_t)gelu_new(val);
        } else {
          resid32[(size_t)row * N + col] += val;
        }
      }
    }
  }
}

// ---------------------------------------------------------------------------
// Fused ReLU-attention:  a = relu(scale * Q K^T) V  per (b,h).
// Block = one (b,h) and 128 query rows; 8 waves each own 16 rows.  Loop over
// 32-wide KV tiles: S-tile via 2 WMMA k-steps, scale+relu, round-trip through
// per-wave LDS scratch (C-layout -> A-layout), then 4 WMMAs into the 16x64
// output accumulator.  No scores buffer ever touches HBM.
// ---------------------------------------------------------------------------
__global__ __launch_bounds__(256) void attn_relu_k(
    const half_t* __restrict__ qkv, half_t* __restrict__ aout) {
  __shared__ half_t Qs[128 * 72];       // [n][d] d-pitch 72
  __shared__ half_t Ks[32 * 72];        // [j][d]
  __shared__ half_t Vs[64 * 40];        // [d][j]  (transposed V tile)
  __shared__ half_t Ss[8 * 16 * 40];    // per-wave 16x32 S scratch

  int tid = threadIdx.x, lane = tid & 31, w = tid >> 5;
  int r = lane & 15, hi8 = (lane >> 4) << 3;

  int bh = blockIdx.x;                  // 0..127
  int b = bh >> 4, hh = bh & 15;
  int q0 = blockIdx.y * 128;
  const half_t* base = qkv + (size_t)(b * SEQ) * (3 * DMODEL) + hh * HD;

  // Q tile (loaded once)
#pragma unroll
  for (int c = 0; c < 4; ++c) {
    int cid = c * 256 + tid;
    int qr = cid >> 3, qc = (cid & 7) << 3;
    *(v8h*)&Qs[qr * 72 + qc] =
        *(const v8h*)(base + (size_t)(q0 + qr) * (3 * DMODEL) + qc);
  }

  const v8f vzero = {0.f, 0.f, 0.f, 0.f, 0.f, 0.f, 0.f, 0.f};
  v8f aacc[4] = {vzero, vzero, vzero, vzero};
  const float scale = 0.125f;           // 1/sqrt(HD)
  half_t* sw = &Ss[w * 16 * 40];

  for (int j0 = 0; j0 < SEQ; j0 += 32) {
    // K tile [j][d]
    {
      int kr = tid >> 3, kc = (tid & 7) << 3;
      *(v8h*)&Ks[kr * 72 + kc] =
          *(const v8h*)(base + DMODEL + (size_t)(j0 + kr) * (3 * DMODEL) + kc);
    }
    // V tile transposed -> Vs[d][j]
    {
      int vj = tid & 31, vd0 = (tid >> 5) << 3;
      v8h vv = *(const v8h*)(base + 2 * DMODEL +
                             (size_t)(j0 + vj) * (3 * DMODEL) + vd0);
#pragma unroll
      for (int i = 0; i < 8; ++i) Vs[(vd0 + i) * 40 + vj] = vv[i];
    }
    __syncthreads();

    // GEMM1: S[16x32] = Q[16x64] @ K^T, K-dim=HD=64 -> 2 WMMA k-steps
    v8f s[2] = {vzero, vzero};
    const half_t* qrow = &Qs[(w * 16 + r) * 72];
#pragma unroll
    for (int kk = 0; kk < 64; kk += 32) {
      v16h aq  = frag_ld(qrow + kk, hi8);
      v16h bk0 = frag_ld(&Ks[(r) * 72 + kk], hi8);
      v16h bk1 = frag_ld(&Ks[(16 + r) * 72 + kk], hi8);
      s[0] = wmma_f16(aq, bk0, s[0]);
      s[1] = wmma_f16(aq, bk1, s[1]);
    }
    // scale + relu, C-layout -> A-layout via this wave's LDS scratch
#pragma unroll
    for (int ni = 0; ni < 2; ++ni)
#pragma unroll
      for (int v = 0; v < 8; ++v)
        sw[(v + hi8) * 40 + ni * 16 + r] =
            (half_t)fmaxf(s[ni][v] * scale, 0.0f);

    // GEMM2: aacc[16x64] += S16[16x32] @ V[32x64]
    v16h as = frag_ld(&sw[r * 40], hi8);
#pragma unroll
    for (int di = 0; di < 4; ++di) {
      v16h bv = frag_ld(&Vs[(di * 16 + r) * 40], hi8);
      aacc[di] = wmma_f16(as, bv, aacc[di]);
    }
    __syncthreads();
  }

  // Store a (f16, token-major [4096][1024]) for the attn-out GEMM.
  half_t* ob = aout + (size_t)(b * SEQ + q0 + w * 16) * DMODEL + hh * HD;
#pragma unroll
  for (int di = 0; di < 4; ++di)
#pragma unroll
    for (int v = 0; v < 8; ++v)
      ob[(size_t)(v + hi8) * DMODEL + di * 16 + r] = (half_t)aacc[di][v];
}

// ---------------------------------------------------------------------------
// Head: out[b] = sum_i hf[b][i] * Wro[i] + bro   (i over N*D = 524288)
// ---------------------------------------------------------------------------
__global__ __launch_bounds__(256) void head_k(
    const float* __restrict__ hf, const float* __restrict__ Wro,
    const float* __restrict__ bro, float* __restrict__ out) {
  __shared__ float red[256];
  int b = blockIdx.x, tid = threadIdx.x;
  const float* hr = hf + (size_t)b * (SEQ * DMODEL);
  float s = 0.f;
  for (int i = tid; i < SEQ * DMODEL; i += 256) s += hr[i] * Wro[i];
  red[tid] = s; __syncthreads();
  for (int off = 128; off; off >>= 1) {
    if (tid < off) red[tid] += red[tid + off];
    __syncthreads();
  }
  if (tid == 0) out[b] = red[0] + bro[0];
}

// ---------------------------------------------------------------------------
extern "C" void kernel_launch(void* const* d_in, const int* in_sizes, int n_in,
                              void* d_out, int out_size, void* d_ws,
                              size_t ws_size, hipStream_t stream) {
  (void)in_sizes; (void)n_in; (void)out_size; (void)ws_size;

  const float* xs    = (const float*)d_in[0];
  const float* Wri   = (const float*)d_in[1];
  const float* bri   = (const float*)d_in[2];
  const float* wpe   = (const float*)d_in[3];
  const float* Wqkv  = (const float*)d_in[4];
  const float* bqkv  = (const float*)d_in[5];
  const float* Wo    = (const float*)d_in[6];
  const float* bo    = (const float*)d_in[7];
  const float* ln1_g = (const float*)d_in[8];
  const float* ln1_b = (const float*)d_in[9];
  const float* ln2_g = (const float*)d_in[10];
  const float* ln2_b = (const float*)d_in[11];
  const float* Wfc   = (const float*)d_in[12];
  const float* bfc   = (const float*)d_in[13];
  const float* Wproj = (const float*)d_in[14];
  const float* bproj = (const float*)d_in[15];
  const float* lnf_g = (const float*)d_in[16];
  const float* lnf_b = (const float*)d_in[17];
  const float* Wro   = (const float*)d_in[18];
  const float* bro   = (const float*)d_in[19];

  // Workspace layout (~104 MB)
  char* ws = (char*)d_ws;
  float*  h      = (float*)(ws);                         // 16,777,216 B
  half_t* x16    = (half_t*)(ws + 16777216);             //  8,388,608 B
  half_t* qkv16  = (half_t*)(ws + 25165824);             // 25,165,824 B
  half_t* a16    = (half_t*)(ws + 50331648);             //  8,388,608 B
  half_t* mid16  = (half_t*)(ws + 58720256);             // 33,554,432 B
  float*  hf     = (float*)(ws + 92274688);              // 16,777,216 B

  embed_k<<<NTOK, 256, 0, stream>>>(xs, Wri, bri, wpe, h);

  for (int l = 0; l < 8; ++l) {
    const float* Wqkv_l  = Wqkv  + (size_t)l * DMODEL * 3 * DMODEL;
    const float* bqkv_l  = bqkv  + (size_t)l * 3 * DMODEL;
    const float* Wo_l    = Wo    + (size_t)l * DMODEL * DMODEL;
    const float* bo_l    = bo    + (size_t)l * DMODEL;
    const float* Wfc_l   = Wfc   + (size_t)l * DMODEL * DFF;
    const float* bfc_l   = bfc   + (size_t)l * DFF;
    const float* Wproj_l = Wproj + (size_t)l * DFF * DMODEL;
    const float* bproj_l = bproj + (size_t)l * DMODEL;

    layernorm_k<half_t><<<NTOK, 256, 0, stream>>>(
        h, ln1_g + l * DMODEL, ln1_b + l * DMODEL, x16);

    gemm_ws<0><<<dim3(3 * DMODEL / 128, NTOK / 128), 256, 0, stream>>>(
        x16, Wqkv_l, bqkv_l, nullptr, qkv16, NTOK, 3 * DMODEL, DMODEL);

    attn_relu_k<<<dim3(8 * NHEAD, SEQ / 128), 256, 0, stream>>>(qkv16, a16);

    gemm_ws<2><<<dim3(DMODEL / 128, NTOK / 128), 256, 0, stream>>>(
        a16, Wo_l, bo_l, h, nullptr, NTOK, DMODEL, DMODEL);

    layernorm_k<half_t><<<NTOK, 256, 0, stream>>>(
        h, ln2_g + l * DMODEL, ln2_b + l * DMODEL, x16);

    gemm_ws<1><<<dim3(DFF / 128, NTOK / 128), 256, 0, stream>>>(
        x16, Wfc_l, bfc_l, nullptr, mid16, NTOK, DFF, DMODEL);

    gemm_ws<2><<<dim3(DMODEL / 128, NTOK / 128), 256, 0, stream>>>(
        mid16, Wproj_l, bproj_l, h, nullptr, NTOK, DMODEL, DFF);
  }

  layernorm_k<float><<<NTOK, 256, 0, stream>>>(h, lnf_g, lnf_b, hf);
  head_k<<<8, 256, 0, stream>>>(hf, Wro, bro, (float*)d_out);
}
